// SNNModel_39840116638557
// MI455X (gfx1250) — matrix operand order repaired and checked by reading, not compiled, not run on hardware
//
#include <hip/hip_runtime.h>

// ---------------- problem constants ----------------
#define N_IN   16384
#define N0     4096
#define N1     1024
#define TSTEPS 599          // T-1 steps (x[1:])
#define STIM_T 600
#define MAXDLY 16
#define NPAD   640          // padded time dim for GEMM tiling (5 * 128)
#define SPS    616          // padded spike-raster row stride (16 zeros + 599 + 1)

// ---------------- GEMM tiling ----------------
#define BM 128
#define BN 128
#define BK 32

typedef __attribute__((ext_vector_type(16))) __bf16 v16bf;
typedef __attribute__((ext_vector_type(8)))  float  v8f;

struct U4x2 { uint4 lo, hi; };

// fp32 pair -> packed bf16 dword.
// Round-half-up (+0x8000) then grab the upper two bytes of each value
// with a single v_perm_b32: byte0,1 <- ra[2],ra[3]; byte2,3 <- rb[2],rb[3].
__device__ __forceinline__ unsigned int pkbf(float a, float b) {
    unsigned int ua = __builtin_bit_cast(unsigned int, a) + 0x8000u;
    unsigned int ub = __builtin_bit_cast(unsigned int, b) + 0x8000u;
#if __has_builtin(__builtin_amdgcn_perm)
    return __builtin_amdgcn_perm(ub, ua, 0x07060302u);
#else
    return (ua >> 16) | (ub & 0xFFFF0000u);
#endif
}

// ============================================================
// Kernel 1: I0 (4096 x NPAD) = W0 (4096 x 16384) @ X (16384 x NPAD)
//           X[k][n] = stim[k*600 + n + 1]  (n < 599, else 0)
// bf16 WMMA 16x16x32, fp32 accumulate. LDS-tiled 128x128x32.
// Grid is (n-tiles, m-tiles) so the 5 blocks sharing a W0 row
// stripe are launch-adjacent -> stripe served from L2 4/5 times.
// ============================================================
__global__ __launch_bounds__(256)
void gemm_I0_kernel(const float* __restrict__ W0,
                    const float* __restrict__ stim,
                    float* __restrict__ I0)
{
    __shared__ __align__(16) unsigned short As[BM * BK]; // row-major, 64B rows
    __shared__ __align__(16) unsigned short Bs[BN * BK]; // col-major, 64B cols

    const int tid  = threadIdx.x;
    const int lane = tid & 31;
    const int wave = tid >> 5;      // 0..7
    const int wm   = wave & 3;      // 4 wave-rows  x 32 rows
    const int wn   = wave >> 2;     // 2 wave-cols  x 64 cols
    const int n0   = blockIdx.x * BN;   // fast grid dim -> time tiles
    const int m0   = blockIdx.y * BM;   // slow grid dim -> W0 row stripes

    // A staging assignment: thread -> one half-row (16 floats)
    const int ra     = tid >> 1;          // 0..127
    const int colseg = (tid & 1) * 16;    // 0 or 16
    const float* gA  = W0 + (size_t)(m0 + ra) * N_IN + colseg;

    // B staging assignment: thread -> one column, 8 packed k-pairs
    const int nloc  = tid & 127;          // local column
    const int khalf = tid >> 7;           // 0/1 -> k-pair groups
    const int ng    = n0 + nloc;
    const bool nvalid = (ng < TSTEPS);

    v8f acc[2][4];
    #pragma unroll
    for (int a = 0; a < 2; ++a)
        #pragma unroll
        for (int b = 0; b < 4; ++b) acc[a][b] = v8f{};

    for (int k0 = 0; k0 < N_IN; k0 += BK) {
        // ---- stage A tile (fp32 -> bf16) ----
        {
            const float* g = gA + k0;
            float4 f0 = *(const float4*)(g + 0);
            float4 f1 = *(const float4*)(g + 4);
            float4 f2 = *(const float4*)(g + 8);
            float4 f3 = *(const float4*)(g + 12);
            unsigned int* dst = (unsigned int*)As + ((ra * BK + colseg) >> 1);
            dst[0] = pkbf(f0.x, f0.y); dst[1] = pkbf(f0.z, f0.w);
            dst[2] = pkbf(f1.x, f1.y); dst[3] = pkbf(f1.z, f1.w);
            dst[4] = pkbf(f2.x, f2.y); dst[5] = pkbf(f2.z, f2.w);
            dst[6] = pkbf(f3.x, f3.y); dst[7] = pkbf(f3.z, f3.w);
            if (k0 + BK < N_IN) __builtin_prefetch(gA + k0 + BK, 0, 1); // global_prefetch_b8
        }
        // ---- stage B tile (gathered from stim, fp32 -> bf16, col-major) ----
        {
            unsigned int* dst = (unsigned int*)Bs + nloc * (BK / 2);
            #pragma unroll
            for (int j = 0; j < 8; ++j) {
                int p = khalf * 8 + j;            // k-pair index 0..15
                int k = k0 + 2 * p;
                float f0 = 0.0f, f1 = 0.0f;
                if (nvalid) {
                    f0 = stim[(size_t)k * STIM_T + ng + 1];
                    f1 = stim[(size_t)(k + 1) * STIM_T + ng + 1];
                }
                dst[p] = pkbf(f0, f1);
            }
        }
        __syncthreads();

        // ---- load fragments from LDS (layouts per ISA 7.12.2) ----
        v16bf afrag[2];
        #pragma unroll
        for (int mt = 0; mt < 2; ++mt) {
            int row  = wm * 32 + mt * 16 + (lane & 15);
            int base = row * (BK * 2) + (lane >> 4) * 16;   // bytes
            U4x2 t;
            t.lo = *(const uint4*)((const char*)As + base);
            t.hi = *(const uint4*)((const char*)As + base + 32);
            afrag[mt] = __builtin_bit_cast(v16bf, t);
        }
        v16bf bfrag[4];
        #pragma unroll
        for (int nt = 0; nt < 4; ++nt) {
            int col  = wn * 64 + nt * 16 + (lane & 15);
            int base = col * (BK * 2) + (lane >> 4) * 16;   // bytes
            U4x2 t;
            t.lo = *(const uint4*)((const char*)Bs + base);
            t.hi = *(const uint4*)((const char*)Bs + base + 32);
            bfrag[nt] = __builtin_bit_cast(v16bf, t);
        }

        // ---- 8 WMMAs per wave per k-step ----
        #pragma unroll
        for (int mt = 0; mt < 2; ++mt)
            #pragma unroll
            for (int nt = 0; nt < 4; ++nt)
                acc[mt][nt] = __builtin_amdgcn_wmma_f32_16x16x32_bf16(
                    false, afrag[mt], false, bfrag[nt],
                    (short)0, acc[mt][nt], false, false);

        __syncthreads();
    }

    // ---- epilogue: scatter C tile to I0 (padded NPAD stride) ----
    #pragma unroll
    for (int mt = 0; mt < 2; ++mt) {
        #pragma unroll
        for (int nt = 0; nt < 4; ++nt) {
            int row0 = m0 + wm * 32 + mt * 16 + (lane >> 4) * 8;
            int col  = n0 + wn * 64 + nt * 16 + (lane & 15);
            #pragma unroll
            for (int v = 0; v < 8; ++v)
                I0[(size_t)(row0 + v) * NPAD + col] = acc[mt][nt][v];
        }
    }
}

// ============================================================
// Kernel 2: layer-0 LIF scan. One thread per neuron.
// Writes spikes to d_out (S0, [4096 x 599]) and to the padded
// raster s0p ([4096 x SPS], 16 leading zeros per row).
// ============================================================
__global__ __launch_bounds__(256)
void lif0_kernel(const float* __restrict__ I0,
                 float* __restrict__ s0_out,
                 float* __restrict__ s0p)
{
    int m = blockIdx.x * blockDim.x + threadIdx.x;
    if (m >= N0) return;
    const float* Irow = I0 + (size_t)m * NPAD;
    float* so = s0_out + (size_t)m * TSTEPS;
    float* sp = s0p + (size_t)m * SPS;
    #pragma unroll
    for (int d = 0; d < MAXDLY; ++d) sp[d] = 0.0f;
    float v = 0.0f;
    for (int t = 0; t < TSTEPS; ++t) {
        v = v * 0.9f + Irow[t];
        float s = (v >= 1.0f) ? 1.0f : 0.0f;
        so[t] = s;
        sp[MAXDLY + t] = s;
        v = (s != 0.0f) ? 0.0f : v;
    }
}

// ============================================================
// Kernel 3: layer-1 delayed gather + LIF. One block per output
// neuron j. W1/delays row staged in LDS (32 KB); threads cover
// time steps; single-lane scan at the end. s0p (~10 MB) and
// W1+delays (32 MB) are L2-resident.
// ============================================================
__global__ __launch_bounds__(256)
void layer1_kernel(const float* __restrict__ W1,
                   const int* __restrict__ delays,
                   const float* __restrict__ s0p,
                   float* __restrict__ out1)
{
    __shared__ float w[N0];
    __shared__ int   dl[N0];
    __shared__ float I1s[768];

    const int j   = blockIdx.x;
    const int tid = threadIdx.x;

    for (int i = tid; i < N0; i += 256) {
        w[i]  = W1[(size_t)j * N0 + i];
        dl[i] = delays[(size_t)j * N0 + i];
    }
    __syncthreads();

    const int t0 = tid, t1 = tid + 256, t2 = tid + 512;
    const int c0 = min(t0, TSTEPS - 1);
    const int c1 = min(t1, TSTEPS - 1);
    const int c2 = min(t2, TSTEPS - 1);

    float a0 = 0.0f, a1 = 0.0f, a2 = 0.0f;
    for (int i = 0; i < N0; ++i) {
        float wi = w[i];
        int   di = dl[i];
        const float* row = s0p + (size_t)i * SPS + MAXDLY - di;
        a0 += wi * row[c0];
        a1 += wi * row[c1];
        a2 += wi * row[c2];
    }
    I1s[t0] = a0;
    I1s[t1] = a1;
    I1s[t2] = a2;
    __syncthreads();

    if (tid == 0) {
        float v = 0.0f;
        float* o = out1 + (size_t)j * TSTEPS;
        for (int t = 0; t < TSTEPS; ++t) {
            v = v * 0.9f + I1s[t];
            float s = (v >= 1.0f) ? 1.0f : 0.0f;
            o[t] = s;
            v = (s != 0.0f) ? 0.0f : v;
        }
    }
}

// ============================================================
extern "C" void kernel_launch(void* const* d_in, const int* in_sizes, int n_in,
                              void* d_out, int out_size, void* d_ws, size_t ws_size,
                              hipStream_t stream) {
    (void)in_sizes; (void)n_in; (void)out_size; (void)ws_size;
    const float* stim   = (const float*)d_in[0];   // (128,128,600)
    const float* W0     = (const float*)d_in[1];   // (4096,16384)
    const float* W1     = (const float*)d_in[2];   // (1024,4096)
    const int*   delays = (const int*)d_in[3];     // (1024,4096)
    float* out = (float*)d_out;

    float* I0  = (float*)d_ws;                     // 4096*640 f32
    float* s0p = I0 + (size_t)N0 * NPAD;           // 4096*616 f32

    dim3 g1(NPAD / BN, N0 / BM);                   // (5, 32): n fast, m slow
    gemm_I0_kernel<<<g1, 256, 0, stream>>>(W0, stim, I0);
    lif0_kernel<<<N0 / 256, 256, 0, stream>>>(I0, out, s0p);
    layer1_kernel<<<N1, 256, 0, stream>>>(W1, delays, s0p,
                                          out + (size_t)N0 * TSTEPS);
}
